// skel_net_44710609551764
// MI455X (gfx1250) — matrix operand order repaired and checked by reading, not compiled, not run on hardware
//
#include <hip/hip_runtime.h>
#include <hip/hip_bf16.h>

typedef __attribute__((ext_vector_type(16))) _Float16 v16h;
typedef __attribute__((ext_vector_type(8)))  _Float16 v8h;
typedef __attribute__((ext_vector_type(8)))  float    v8f;

// ---------------------------------------------------------------- constants
#define BN   16384          // batch
#define PP   17             // points per sample
#define M2   (BN * PP)      // 278528 rows after reshape

// workspace layout (all section sizes are multiples of 256 bytes)
#define OFF_W2H   ((size_t)0)                              // 128x64 f16
#define OFF_W3H   (OFF_W2H  + (size_t)128*64*2)            // 1024x128 f16
#define OFF_FW1H  (OFF_W3H  + (size_t)1024*128*2)          // 512x1024 f16
#define OFF_FW2H  (OFF_FW1H + (size_t)512*1024*2)          // 256x512 f16
#define OFF_FW3H  (OFF_FW2H + (size_t)256*512*2)           // 79x256 f16
#define OFF_H2    (OFF_FW3H + (size_t)79*256*2)            // M2 x 128 f16
#define OFF_G     (OFF_H2   + (size_t)M2*128*2)            // BN x 1024 f16
#define OFF_A1    (OFF_G    + (size_t)BN*1024*2)           // BN x 512 f16
#define OFF_A2    (OFF_A1   + (size_t)BN*512*2)            // BN x 256 f16
#define OFF_OUTQ  (OFF_A2   + (size_t)BN*256*2)            // BN x 80 f32

// ---------------------------------------------------------------- helpers
__device__ __forceinline__ v8f zero8f() {
    v8f z;
    #pragma unroll
    for (int i = 0; i < 8; ++i) z[i] = 0.0f;
    return z;
}
__device__ __forceinline__ v16h zero16h() {
    v16h z;
    #pragma unroll
    for (int i = 0; i < 16; ++i) z[i] = (_Float16)0.0f;
    return z;
}

// A fragment (16x32 f16): lane holds row m=lane&15; halves [0..7]=K[k0..k0+7],
// halves [8..15]=K[k0+16..k0+23], k0 = 8*(lane>=16).  p points at row + kstep*32.
__device__ __forceinline__ v16h load_a_frag(const _Float16* p, int k0) {
    v8h lo = *(const v8h*)(p + k0);
    v8h hi = *(const v8h*)(p + k0 + 16);
    v16h f;
    #pragma unroll
    for (int i = 0; i < 8; ++i) { f[i] = lo[i]; f[i + 8] = hi[i]; }
    return f;
}

// B fragment (32x16 f16): lane holds weight-row n=lane&15 (W is NxK row-major),
// halves [0..7]=K[k0..k0+7], halves [8..15]=K[k0+16..k0+23], k0 = 8*(lane>=16).
__device__ __forceinline__ v16h load_b_frag(const _Float16* p, int hi) {
    v8h lo = *(const v8h*)(p + hi * 8);
    v8h up = *(const v8h*)(p + hi * 8 + 16);
    v16h f;
    #pragma unroll
    for (int i = 0; i < 8; ++i) { f[i] = lo[i]; f[i + 8] = up[i]; }
    return f;
}

__device__ __forceinline__ v8f wmma16(v16h a, v16h b, v8f c) {
    return __builtin_amdgcn_wmma_f32_16x16x32_f16(
        /*neg_a=*/false, a, /*neg_b=*/false, b,
        /*c_mod=*/(short)0, c, /*reuse_a=*/false, /*reuse_b=*/false);
}

// ---------------------------------------------------------------- kernels
__global__ void cvt_f16_kernel(const float* __restrict__ src,
                               _Float16* __restrict__ dst, int n) {
    int i = blockIdx.x * blockDim.x + threadIdx.x;
    for (; i < n; i += gridDim.x * blockDim.x) dst[i] = (_Float16)src[i];
}

// stage 1: per-row MLP 2->64 (VALU) then 64->128 (WMMA), relu, H2 (M2 x 128) f16.
// block = 128 threads (4 waves); wave w handles rows [ (bx*4+w)*16 .. +15 ].
__global__ void stage1_kernel(const float* __restrict__ x,
                              const float* __restrict__ w1,   // (64,2)
                              const float* __restrict__ b1,   // (64)
                              const _Float16* __restrict__ w2h, // (128,64)
                              const float* __restrict__ b2,   // (128)
                              _Float16* __restrict__ h2) {
    __shared__ float sw1[128];
    __shared__ float sb1[64];
    __shared__ float sb2[128];
    int t = threadIdx.x;
    sw1[t] = w1[t];
    if (t < 64) sb1[t] = b1[t];
    sb2[t] = b2[t];
    __syncthreads();

    int lane = t & 31, wave = t >> 5;
    int m = lane & 15, hi = lane >> 4, k0 = hi * 8;
    int rbase = (blockIdx.x * 4 + wave) * 16;
    int r = rbase + m;
    int b = r / PP, p = r - b * PP;
    float x0 = x[b * 34 + p];
    float x1 = x[b * 34 + 17 + p];

    // build A fragments (layer-1 output, 16 rows x 64 ch) for 2 K-steps
    v16h afr[2];
    #pragma unroll
    for (int ks = 0; ks < 2; ++ks) {
        #pragma unroll
        for (int j = 0; j < 16; ++j) {
            int k = ks * 32 + k0 + (j < 8 ? j : j + 8);
            float h = fmaxf(sw1[2 * k] * x0 + sw1[2 * k + 1] * x1 + sb1[k], 0.0f);
            afr[ks][j] = (_Float16)h;
        }
    }

    #pragma unroll
    for (int nt = 0; nt < 8; ++nt) {
        const _Float16* wrow = w2h + (size_t)(nt * 16 + m) * 64;
        v8f acc = zero8f();
        acc = wmma16(afr[0], load_b_frag(wrow, hi), acc);
        acc = wmma16(afr[1], load_b_frag(wrow + 32, hi), acc);
        int n = nt * 16 + m;
        float bb = sb2[n];
        #pragma unroll
        for (int i = 0; i < 8; ++i) {
            int rr = rbase + i + 8 * hi;
            h2[(size_t)rr * 128 + n] = (_Float16)fmaxf(acc[i] + bb, 0.0f);
        }
    }
}

// stage 2: L3 (128 -> 1024) fused with max over the 17 points.
// grid = (BN/16, 64/4); block 128 = 4 waves; wave w: o-tile = by*4+w, b-tile = bx.
__global__ void stage2_kernel(const _Float16* __restrict__ h2,   // (M2,128)
                              const _Float16* __restrict__ w3h,  // (1024,128)
                              const float* __restrict__ b3,      // (1024)
                              _Float16* __restrict__ g) {        // (BN,1024)
    int lane = threadIdx.x & 31, wave = threadIdx.x >> 5;
    int m = lane & 15, hi = lane >> 4, k0 = hi * 8;
    int bt = blockIdx.x;
    int ot = blockIdx.y * 4 + wave;
    int n = ot * 16 + m;

    v16h bf[4];
    #pragma unroll
    for (int ks = 0; ks < 4; ++ks)
        bf[ks] = load_b_frag(w3h + (size_t)n * 128 + ks * 32, hi);
    float bb = b3[n];

    float gmax[8];
    #pragma unroll
    for (int i = 0; i < 8; ++i) gmax[i] = 0.0f;   // relu output >= 0

    for (int p = 0; p < PP; ++p) {
        const _Float16* arow = h2 + ((size_t)(bt * 16 + m) * PP + p) * 128;
        v8f acc = zero8f();
        #pragma unroll
        for (int ks = 0; ks < 4; ++ks)
            acc = wmma16(load_a_frag(arow + ks * 32, k0), bf[ks], acc);
        #pragma unroll
        for (int i = 0; i < 8; ++i)
            gmax[i] = fmaxf(gmax[i], fmaxf(acc[i] + bb, 0.0f));
    }
    #pragma unroll
    for (int i = 0; i < 8; ++i)
        g[(size_t)(bt * 16 + i + 8 * hi) * 1024 + n] = (_Float16)gmax[i];
}

// generic FC: out = act(A @ W^T + bias). A:(16384,K) f16, W:(N,K) f16 row-major.
// grid = (1024/4, ceil(N/16)); block 128 = 4 waves.
template <bool RELU, bool OUTF16>
__global__ void fc_kernel(const _Float16* __restrict__ A,
                          const _Float16* __restrict__ W,
                          const float* __restrict__ bias,
                          void* __restrict__ out,
                          int K, int Nreal, int Nstore) {
    int lane = threadIdx.x & 31, wave = threadIdx.x >> 5;
    int m = lane & 15, hi = lane >> 4, k0 = hi * 8;
    int mt = blockIdx.x * 4 + wave;
    int nt = blockIdx.y;
    int ncol = nt * 16 + m;

    const _Float16* arow = A + (size_t)(mt * 16 + m) * K;
    const _Float16* wrow = (ncol < Nreal) ? (W + (size_t)ncol * K) : (const _Float16*)0;

    v8f acc = zero8f();
    int ksteps = K >> 5;
    for (int ks = 0; ks < ksteps; ++ks) {
        __builtin_prefetch(arow + (ks + 2) * 32, 0, 0);   // global_prefetch_b8
        v16h af = load_a_frag(arow + ks * 32, k0);
        v16h bfv = zero16h();
        if (wrow) bfv = load_b_frag(wrow + ks * 32, hi);
        acc = wmma16(af, bfv, acc);
    }

    if (ncol < Nreal) {
        float bb = bias[ncol];
        #pragma unroll
        for (int i = 0; i < 8; ++i) {
            int mr = mt * 16 + i + 8 * hi;
            float v = acc[i] + bb;
            if (RELU) v = fmaxf(v, 0.0f);
            if (OUTF16) ((_Float16*)out)[(size_t)mr * Nstore + ncol] = (_Float16)v;
            else        ((float*)out)[(size_t)mr * Nstore + ncol] = v;
        }
    }
}

// forward kinematics + projection: one thread per sample (negligible FLOPs).
__global__ void fk_kernel(const float* __restrict__ outq,  // (BN,80), 79 valid
                          float* __restrict__ out) {
    int tid = blockIdx.x * blockDim.x + threadIdx.x;
    if (tid >= BN) return;
    const float* q = outq + (size_t)tid * 80;

    const int PAR[19] = {-1, 0, 1, 2, 3, 0, 5, 6, 7, 0, 9, 10, 11, 8, 13, 14, 8, 16, 17};
    float rotg[19][9];
    float pos[19][3];

    #pragma unroll
    for (int j = 0; j < 19; ++j) {
        float w = q[3 + 4 * j + 0], xx = q[3 + 4 * j + 1];
        float yy = q[3 + 4 * j + 2], zz = q[3 + 4 * j + 3];
        float inv = 1.0f / (sqrtf(w * w + xx * xx + yy * yy + zz * zz) + 1e-8f);
        w *= inv; xx *= inv; yy *= inv; zz *= inv;
        float R[9] = {
            1.0f - 2.0f * (yy * yy + zz * zz), 2.0f * (xx * yy - w * zz), 2.0f * (xx * zz + w * yy),
            2.0f * (xx * yy + w * zz), 1.0f - 2.0f * (xx * xx + zz * zz), 2.0f * (yy * zz - w * xx),
            2.0f * (xx * zz - w * yy), 2.0f * (yy * zz + w * xx), 1.0f - 2.0f * (xx * xx + yy * yy)};
        if (j == 0) {
            pos[0][0] = q[0]; pos[0][1] = q[1]; pos[0][2] = q[2];
            #pragma unroll
            for (int k = 0; k < 9; ++k) rotg[0][k] = R[k];
        } else {
            int pj = PAR[j];
            float offy = 0.2f + 0.4f * (float)(j - 1) * (1.0f / 17.0f);
            pos[j][0] = pos[pj][0] + rotg[pj][1] * offy;   // offset = (0, offy, 0)
            pos[j][1] = pos[pj][1] + rotg[pj][4] * offy;
            pos[j][2] = pos[pj][2] + rotg[pj][7] * offy;
            #pragma unroll
            for (int r = 0; r < 3; ++r)
                #pragma unroll
                for (int c = 0; c < 3; ++c)
                    rotg[j][r * 3 + c] = rotg[pj][r * 3 + 0] * R[0 + c] +
                                         rotg[pj][r * 3 + 1] * R[3 + c] +
                                         rotg[pj][r * 3 + 2] * R[6 + c];
        }
    }

    // fake_position = P / (1 + eps)
    float* out2 = out + (size_t)BN * 34 + (size_t)tid * 57;
    const float s = 1.0f / (1.0f + 1e-9f);
    #pragma unroll
    for (int j = 0; j < 19; ++j) {
        out2[j * 3 + 0] = pos[j][0] * s;
        out2[j * 3 + 1] = pos[j][1] * s;
        out2[j * 3 + 2] = pos[j][2] * s;
    }
    // pose2d: cam shifts y by -100, project, divide by z, drop joints 9 & 10
    float* out1 = out + (size_t)tid * 34;
    #pragma unroll
    for (int sidx = 0; sidx < 17; ++sidx) {
        int j = (sidx < 9) ? sidx : sidx + 2;
        float invz = 1.0f / (pos[j][2] + 1e-9f);
        out1[sidx * 2 + 0] = pos[j][0] * invz;
        out1[sidx * 2 + 1] = (pos[j][1] - 100.0f) * invz;
    }
}

// ---------------------------------------------------------------- launcher
extern "C" void kernel_launch(void* const* d_in, const int* in_sizes, int n_in,
                              void* d_out, int out_size, void* d_ws, size_t ws_size,
                              hipStream_t stream) {
    const float* x   = (const float*)d_in[0];
    const float* w1  = (const float*)d_in[1];
    const float* b1  = (const float*)d_in[2];
    const float* w2  = (const float*)d_in[3];
    const float* b2  = (const float*)d_in[4];
    const float* w3  = (const float*)d_in[5];
    const float* b3  = (const float*)d_in[6];
    const float* fw1 = (const float*)d_in[7];
    const float* fb1 = (const float*)d_in[8];
    const float* fw2 = (const float*)d_in[9];
    const float* fb2 = (const float*)d_in[10];
    const float* fw3 = (const float*)d_in[11];
    const float* fb3 = (const float*)d_in[12];

    char* ws = (char*)d_ws;
    _Float16* w2h  = (_Float16*)(ws + OFF_W2H);
    _Float16* w3h  = (_Float16*)(ws + OFF_W3H);
    _Float16* fw1h = (_Float16*)(ws + OFF_FW1H);
    _Float16* fw2h = (_Float16*)(ws + OFF_FW2H);
    _Float16* fw3h = (_Float16*)(ws + OFF_FW3H);
    _Float16* h2   = (_Float16*)(ws + OFF_H2);
    _Float16* g    = (_Float16*)(ws + OFF_G);
    _Float16* a1   = (_Float16*)(ws + OFF_A1);
    _Float16* a2   = (_Float16*)(ws + OFF_A2);
    float*    outq = (float*)(ws + OFF_OUTQ);

    // fp32 -> fp16 weight conversion (L2-resident afterwards)
    cvt_f16_kernel<<<dim3(32), 256, 0, stream>>>(w2, w2h, 128 * 64);
    cvt_f16_kernel<<<dim3(512), 256, 0, stream>>>(w3, w3h, 1024 * 128);
    cvt_f16_kernel<<<dim3(2048), 256, 0, stream>>>(fw1, fw1h, 512 * 1024);
    cvt_f16_kernel<<<dim3(512), 256, 0, stream>>>(fw2, fw2h, 256 * 512);
    cvt_f16_kernel<<<dim3(80), 256, 0, stream>>>(fw3, fw3h, 79 * 256);

    // stage 1: rows = 278528, 64 rows per block
    stage1_kernel<<<dim3(M2 / 64), 128, 0, stream>>>(x, w1, b1, w2h, b2, h2);

    // stage 2: L3 + max over 17 points
    stage2_kernel<<<dim3(BN / 16, 16), 128, 0, stream>>>(h2, w3h, b3, g);

    // FC head
    fc_kernel<true, true><<<dim3(BN / 64, 512 / 16), 128, 0, stream>>>(
        g, fw1h, fb1, a1, 1024, 512, 512);
    fc_kernel<true, true><<<dim3(BN / 64, 256 / 16), 128, 0, stream>>>(
        a1, fw2h, fb2, a2, 512, 256, 256);
    fc_kernel<false, false><<<dim3(BN / 64, 5), 128, 0, stream>>>(
        a2, fw3h, fb3, outq, 256, 79, 80);

    // FK + projection -> d_out
    fk_kernel<<<dim3(BN / 256), 256, 0, stream>>>(outq, (float*)d_out);
}
